// LearnedSortOrder_v3_34376918237595
// MI455X (gfx1250) — matrix operand order repaired and checked by reading, not compiled, not run on hardware
//
#include <hip/hip_runtime.h>
#include <hip/hip_bf16.h>

typedef __attribute__((ext_vector_type(16))) _Float16 v16h;
typedef __attribute__((ext_vector_type(8)))  float    v8f;

#define SPAN 4096u   // elements per LDS bitonic tile (32KB of LDS)
#define CHUNK 256    // elements per PAV chunk

// ---------- async global<->LDS support (gfx1250) ----------
#if defined(__AMDGCN__) && __has_builtin(__builtin_amdgcn_global_load_async_to_lds_b64) && __has_builtin(__builtin_amdgcn_s_wait_asynccnt)
#define ASYNC_LDS_LOAD 1
#endif
#if defined(__AMDGCN__) && __has_builtin(__builtin_amdgcn_global_store_async_from_lds_b64) && __has_builtin(__builtin_amdgcn_s_wait_asynccnt)
#define ASYNC_LDS_STORE 1
#endif

#if defined(ASYNC_LDS_LOAD) || defined(ASYNC_LDS_STORE)
// builtin signature (from hipcc diagnostic): param0 = v2i __device__(AS1)*, 64-bit payload
typedef int v2i __attribute__((vector_size(8)));
typedef __attribute__((address_space(1))) v2i as1_v2i;
typedef __attribute__((address_space(3))) v2i as3_v2i;
__device__ __forceinline__ as1_v2i* glob_v2i(const void* p) {
    return (as1_v2i*)(unsigned long long)p;
}
__device__ __forceinline__ as3_v2i* ldsp_v2i(void* p) {
    // generic LDS pointer carries the LDS byte offset in its low 32 bits
    return (as3_v2i*)(unsigned int)(unsigned long long)p;
}
#endif

// ---------- helpers ----------
__device__ __forceinline__ unsigned int ordered_u32(float f) {
    unsigned int b = __float_as_uint(f);
    return (b & 0x80000000u) ? ~b : (b | 0x80000000u);
}
__device__ __forceinline__ float decode_score(unsigned long long key) {
    unsigned int o = ~(unsigned int)(key >> 32);
    unsigned int b = (o & 0x80000000u) ? (o & 0x7fffffffu) : ~o;
    return __uint_as_float(b);
}
__device__ __forceinline__ float logaddexpf_(float a, float b) {
    float mx = fmaxf(a, b), mn = fminf(a, b);
    return mx + log1pf(expf(mn - mx));
}

// ---------- 1) MLP scores via WMMA (16 rows / wave, f32 accumulate) ----------
__global__ void mlp_scores_wmma(const float* __restrict__ x,
                                const float* __restrict__ w1,
                                const float* __restrict__ b1,
                                const float* __restrict__ w2,
                                const float* __restrict__ b2,
                                float* __restrict__ scores,
                                int n, int ntiles) {
    // scalar tail (rows beyond ntiles*16); no WMMA inside this divergent branch
    if (blockIdx.x == 0) {
        int tail0 = ntiles * 16;
        int tid = (int)threadIdx.x;
        if (tid < 16 && tail0 + tid < n) {
            float xv = x[tail0 + tid];
            float acc = b2[0];
            for (int k = 0; k < 32; ++k) {
                float h = fmaf(xv, w1[k], b1[k]);
                h = h > 0.f ? h : 0.f;
                acc = fmaf(h, w2[k], acc);
            }
            scores[tail0 + tid] = acc;
        }
    }
    int wave = (int)((blockIdx.x * blockDim.x + threadIdx.x) >> 5);
    int lane = (int)(threadIdx.x & 31u);
    int hf   = lane >> 4;      // K-half selector
    int m    = lane & 15;      // row within tile / column of D
    if (wave >= ntiles) return;          // whole wave exits together -> EXEC all-1 at WMMA

    // B (32x16 f16): element e -> K = e + 8*hf (+8 if e>=8) (w2 replicated per column)
    v16h bv;
#pragma unroll
    for (int e = 0; e < 16; ++e) bv[e] = (_Float16)w2[e + 8 * hf + (e >= 8 ? 8 : 0)];

    // A (16x32 f16): lane row m; element e -> K = e + 8*hf (+8 if e>=8)
    float xv = x[wave * 16 + m];
    v16h av;
#pragma unroll
    for (int e = 0; e < 16; ++e) {
        int k = e + 8 * hf + (e >= 8 ? 8 : 0);
        float h = fmaf(xv, w1[k], b1[k]);
        av[e] = (_Float16)(h > 0.f ? h : 0.f);
    }
    v8f c = {};
    c = __builtin_amdgcn_wmma_f32_16x16x32_f16(false, av, false, bv,
                                               (short)0, c, false, false);
    // D column 0 lives in lanes with m==0: lane 0 -> M=0..7, lane 16 -> M=8..15
    if (m == 0) {
        float bb = b2[0];
#pragma unroll
        for (int r = 0; r < 8; ++r) scores[wave * 16 + 8 * hf + r] = c[r] + bb;
    }
}

// ---------- 2) build 64-bit sort keys: (~ordered(score))<<32 | index ----------
__global__ void build_keys(const float* __restrict__ scores,
                           unsigned long long* __restrict__ keys,
                           int n, unsigned int m) {
    unsigned int i = blockIdx.x * blockDim.x + threadIdx.x;
    if (i >= m) return;
    if ((int)i < n) {
        unsigned int o = ordered_u32(scores[i]);
        keys[i] = ((unsigned long long)(~o) << 32) | (unsigned long long)i;
    } else {
        keys[i] = 0xFFFFFFFFFFFFFFFFull;   // pad sorts to the end
    }
}

// ---------- bitonic tile staging: async global->LDS when available ----------
__device__ __forceinline__ void tile_load(unsigned long long* __restrict__ keys,
                                          unsigned long long* lds, unsigned int base) {
#ifdef ASYNC_LDS_LOAD
    for (unsigned int i = threadIdx.x; i < SPAN; i += blockDim.x)
        __builtin_amdgcn_global_load_async_to_lds_b64(glob_v2i(keys + base + i),
                                                      ldsp_v2i(&lds[i]), 0, 0);
    __builtin_amdgcn_s_wait_asynccnt(0);
    __syncthreads();
#else
    for (unsigned int i = threadIdx.x; i < SPAN; i += blockDim.x) lds[i] = keys[base + i];
    __syncthreads();
#endif
}
__device__ __forceinline__ void tile_store(unsigned long long* __restrict__ keys,
                                           unsigned long long* lds, unsigned int base) {
#ifdef ASYNC_LDS_STORE
    for (unsigned int i = threadIdx.x; i < SPAN; i += blockDim.x)
        __builtin_amdgcn_global_store_async_from_lds_b64(glob_v2i(keys + base + i),
                                                         ldsp_v2i(&lds[i]), 0, 0);
    __builtin_amdgcn_s_wait_asynccnt(0);   // drain before wave exit (ENDPGM also waits)
#else
    for (unsigned int i = threadIdx.x; i < SPAN; i += blockDim.x) keys[base + i] = lds[i];
#endif
}

// ---------- 3) bitonic sort (ascending) ----------
__global__ void bitonic_local_sort(unsigned long long* __restrict__ keys) {
    __shared__ unsigned long long lds[SPAN];
    unsigned int base = blockIdx.x * SPAN;
    tile_load(keys, lds, base);
    for (unsigned int k = 2; k <= SPAN; k <<= 1) {
        for (unsigned int j = k >> 1; j >= 1; j >>= 1) {
            for (unsigned int t = threadIdx.x; t < SPAN / 2; t += blockDim.x) {
                unsigned int i = (t / j) * (2 * j) + (t % j);
                unsigned int gi = base + i;
                bool up = ((gi & k) == 0);
                unsigned long long a = lds[i], b = lds[i + j];
                if ((a > b) == up) { lds[i] = b; lds[i + j] = a; }
            }
            __syncthreads();
        }
    }
    tile_store(keys, lds, base);
}

__global__ void bitonic_global_pass(unsigned long long* __restrict__ keys,
                                    unsigned int j, unsigned int k) {
    unsigned int t = blockIdx.x * blockDim.x + threadIdx.x;
    unsigned int i = (t / j) * (2 * j) + (t % j);
    unsigned int p = i + j;
    bool up = ((i & k) == 0);
    unsigned long long a = keys[i], b = keys[p];
    if ((a > b) == up) { keys[i] = b; keys[p] = a; }
}

__global__ void bitonic_local_merge(unsigned long long* __restrict__ keys, unsigned int k) {
    __shared__ unsigned long long lds[SPAN];
    unsigned int base = blockIdx.x * SPAN;
    tile_load(keys, lds, base);
    for (unsigned int j = SPAN / 2; j >= 1; j >>= 1) {
        for (unsigned int t = threadIdx.x; t < SPAN / 2; t += blockDim.x) {
            unsigned int i = (t / j) * (2 * j) + (t % j);
            unsigned int gi = base + i;
            bool up = ((gi & k) == 0);
            unsigned long long a = lds[i], b = lds[i + j];
            if ((a > b) == up) { lds[i] = b; lds[i + j] = a; }
        }
        __syncthreads();
    }
    tile_store(keys, lds, base);
}

// ---------- 4) per-chunk KL-PAV (parallel; stack state == isotonic fit of chunk) ----------
__global__ void pav_chunks(const unsigned long long* __restrict__ keys,
                           float* __restrict__ cly, float* __restrict__ clw,
                           float* __restrict__ cval, int* __restrict__ cstart,
                           int* __restrict__ ccount, int n) {
    int c = (int)(blockIdx.x * blockDim.x + threadIdx.x);
    int nchunks = (n + CHUNK - 1) / CHUNK;
    if (c >= nchunks) return;
    int s0 = c * CHUNK;
    int s1 = s0 + CHUNK; if (s1 > n) s1 = n;
    float* ly = cly + s0; float* lw = clw + s0; float* lv = cval + s0; int* st = cstart + s0;
    int p = 0;
    for (int i = s0; i < s1; ++i) {
        if (i + 16 < s1) __builtin_prefetch(&keys[i + 16], 0, 3);
        float y = decode_score(keys[i]);
        float w = logf((float)(n - i));          // w_log[i] = log(n - i)
        ly[p] = y; lw[p] = w; lv[p] = y - w; st[p] = i; ++p;
        while (p >= 2 && lv[p - 2] <= lv[p - 1]) {
            float ny = logaddexpf_(ly[p - 2], ly[p - 1]);
            float nw = logaddexpf_(lw[p - 2], lw[p - 1]);
            ly[p - 2] = ny; lw[p - 2] = nw; lv[p - 2] = ny - nw; --p;
        }
    }
    ccount[c] = p;
}

// ---------- 5) sequential merge of chunk block-lists (one thread) ----------
__global__ void pav_merge(const float* __restrict__ cly, const float* __restrict__ clw,
                          const int* __restrict__ cstart, const int* __restrict__ ccount,
                          float* __restrict__ gly, float* __restrict__ glw,
                          float* __restrict__ gval, int* __restrict__ gstart,
                          int* __restrict__ gptr, int n) {
    if (threadIdx.x != 0 || blockIdx.x != 0) return;
    int nchunks = (n + CHUNK - 1) / CHUNK;
    int p = 0;
    for (int c = 0; c < nchunks; ++c) {
        if (c + 1 < nchunks) {                 // hide latency of the next chunk's block list
            __builtin_prefetch(&cly[(c + 1) * CHUNK], 0, 3);
            __builtin_prefetch(&clw[(c + 1) * CHUNK], 0, 3);
            __builtin_prefetch(&cstart[(c + 1) * CHUNK], 0, 3);
        }
        int cnt = ccount[c];
        const float* ly = cly + c * CHUNK;
        const float* lw = clw + c * CHUNK;
        const int*   st = cstart + c * CHUNK;
        for (int b = 0; b < cnt; ++b) {
            gly[p] = ly[b]; glw[p] = lw[b];
            gval[p] = ly[b] - lw[b]; gstart[p] = st[b]; ++p;
            while (p >= 2 && gval[p - 2] <= gval[p - 1]) {
                float ny = logaddexpf_(gly[p - 2], gly[p - 1]);
                float nw = logaddexpf_(glw[p - 2], glw[p - 1]);
                gly[p - 2] = ny; glw[p - 2] = nw; gval[p - 2] = ny - nw; --p;
            }
        }
    }
    *gptr = p;
}

// ---------- 6) finalize: rank = exp(s - sol), quantize, scatter by inverse perm ----------
__global__ void finalize(const unsigned long long* __restrict__ keys,
                         const float* __restrict__ gval, const int* __restrict__ gstart,
                         const int* __restrict__ gptr, float* __restrict__ out, int n) {
    int i = (int)(blockIdx.x * blockDim.x + threadIdx.x);
    if (i >= n) return;
    int p = *gptr;
    int lo = 0, hi = p - 1;
    while (lo < hi) {                       // largest t with gstart[t] <= i
        int mid = (lo + hi + 1) >> 1;
        if (gstart[mid] <= i) lo = mid; else hi = mid - 1;
    }
    float sol = gval[lo];
    unsigned long long key = keys[i];
    float s = decode_score(key);
    int idx = (int)(key & 0xffffffffu);
    float r = expf(s - sol);
    float q = (r - fmodf(r, 3.0f)) * (1.0f / 3.0f) + 1.0f;   // (r - r%capacity)/capacity + 1
    out[idx] = q;
}

// ---------- launch ----------
extern "C" void kernel_launch(void* const* d_in, const int* in_sizes, int n_in,
                              void* d_out, int out_size, void* d_ws, size_t ws_size,
                              hipStream_t stream) {
    const float* x  = (const float*)d_in[0];
    const float* w1 = (const float*)d_in[1];
    const float* b1 = (const float*)d_in[2];
    const float* w2 = (const float*)d_in[3];
    const float* b2 = (const float*)d_in[4];
    float* out = (float*)d_out;
    int n = in_sizes[0];                       // x is [N,1]

    unsigned int M = SPAN;                     // bitonic size: pow2 >= max(N, SPAN)
    while (M < (unsigned int)n) M <<= 1;
    int nchunks = (n + CHUNK - 1) / CHUNK;

    char* ws = (char*)d_ws;
    size_t off = 0;
    unsigned long long* keys = (unsigned long long*)(ws + off); off += (size_t)M * 8;
    float* cly   = (float*)(ws + off); off += (size_t)n * 4;
    float* clw   = (float*)(ws + off); off += (size_t)n * 4;
    float* cval  = (float*)(ws + off); off += (size_t)n * 4;
    int*   cstart= (int*)(ws + off);   off += (size_t)n * 4;
    int*   ccount= (int*)(ws + off);   off += (size_t)(nchunks + 1) * 4;
    float* gly   = (float*)(ws + off); off += (size_t)n * 4;
    float* glw   = (float*)(ws + off); off += (size_t)n * 4;
    float* gval  = (float*)(ws + off); off += (size_t)n * 4;
    int*   gstart= (int*)(ws + off);   off += (size_t)n * 4;
    int*   gptr  = (int*)(ws + off);   off += 64;
    float* scores = out;                       // reused; finalize overwrites every element

    // 1) MLP scores (WMMA)
    int ntiles = n / 16;
    int mlp_blocks = (ntiles + 7) / 8; if (mlp_blocks < 1) mlp_blocks = 1;
    mlp_scores_wmma<<<mlp_blocks, 256, 0, stream>>>(x, w1, b1, w2, b2, scores, n, ntiles);

    // 2) keys
    build_keys<<<(M + 255) / 256, 256, 0, stream>>>(scores, keys, n, M);

    // 3) bitonic sort
    bitonic_local_sort<<<M / SPAN, 512, 0, stream>>>(keys);
    for (unsigned int k = 2 * SPAN; k <= M; k <<= 1) {
        for (unsigned int j = k >> 1; j >= SPAN; j >>= 1)
            bitonic_global_pass<<<(M / 2 + 255) / 256, 256, 0, stream>>>(keys, j, k);
        bitonic_local_merge<<<M / SPAN, 512, 0, stream>>>(keys, k);
    }

    // 4) chunked PAV, 5) merge, 6) finalize
    pav_chunks<<<(nchunks + 127) / 128, 128, 0, stream>>>(keys, cly, clw, cval, cstart, ccount, n);
    pav_merge<<<1, 32, 0, stream>>>(cly, clw, cstart, ccount, gly, glw, gval, gstart, gptr, n);
    finalize<<<(n + 255) / 256, 256, 0, stream>>>(keys, gval, gstart, gptr, out, n);

    (void)n_in; (void)out_size; (void)ws_size;
}
